// AMMPreActBlock_62904091018003
// MI455X (gfx1250) — matrix operand order repaired and checked by reading, not compiled, not run on hardware
//
#include <hip/hip_runtime.h>
#include <hip/hip_bf16.h>

// ---------------------------------------------------------------------------
// CDNA5 (gfx1250, wave32) AMM PreAct block.
// WMMA (v_wmma_f32_16x16x32_bf16) computes the PQ-encode distance terms
// directly: D = patches[16 x d] * centroids^T[d x 16] + C, with the C operand
// carrying -0.5*||centroid||^2, so dist-argmin == D-argmax with zero post-ops.
// Argmin via padded-LDS transpose + branchless per-lane scan.
// ---------------------------------------------------------------------------

typedef __attribute__((ext_vector_type(16))) __bf16 v16bf;
typedef __attribute__((ext_vector_type(8)))  float  v8f;

__device__ __forceinline__ __bf16 f2bf(float f) {
  union { float f; unsigned int u; } a; a.f = f;
  unsigned int u = a.u + 0x7fffu + ((a.u >> 16) & 1u);   // round-to-nearest-even
  union { unsigned short s; __bf16 h; } b;
  b.s = (unsigned short)(u >> 16);
  return b.h;
}

// ---------------------------------------------------------------------------
// Encode tail. acc = cross - 0.5*||c||^2  (bias fed through WMMA C operand).
// argmin(dist) == argmax(acc). C/D layout: VGPR r, lanes 0-15 -> (M=r,N=lane),
// lanes 16-31 -> (M=r+8, N=lane-16). Scores transit LDS with row stride 17
// dwords (conflict-free both directions); lanes 0-15 scan one row each with
// pure cndmask selects (strict > keeps first occurrence, matching argmin).
// ---------------------------------------------------------------------------
__device__ __forceinline__ void wmma_argmax_store(v16bf A, v16bf Bf, float c2,
                                                  float* __restrict__ sbuf,
                                                  unsigned char* __restrict__ dst) {
  v8f cvec;
  const float bias = -0.5f * c2;
#pragma unroll
  for (int r = 0; r < 8; ++r) cvec[r] = bias;
  v8f acc = __builtin_amdgcn_wmma_f32_16x16x32_bf16(
      /*neg_a=*/false, A, /*neg_b=*/false, Bf,
      /*c_mod=*/(short)0, cvec, /*reuse_a=*/false, /*reuse_b=*/false);
  const int lane = threadIdx.x & 31;
  const int n = lane & 15;
  const int hi = lane >> 4;
#pragma unroll
  for (int r = 0; r < 8; ++r)
    sbuf[(r + hi * 8) * 17 + n] = acc[r];
  __syncthreads();                                  // single-wave block: dscnt fence
  if (lane < 16) {
    const float* row = sbuf + lane * 17;
    float best = row[0];
    int   bi = 0;
#pragma unroll
    for (int j = 1; j < 16; ++j) {
      float v = row[j];
      bool  t = v > best;
      best = t ? v : best;
      bi   = t ? j : bi;
    }
    dst[lane] = (unsigned char)bi;
  }
}

// ---------------------------------------------------------------------------
// 3x3 encode (pad=1). One wave per (b, channel-codebook, 16-location tile).
// Subvector d=9 = the 9 spatial taps of one channel (JAX patches are C-major).
// A: lane m -> K0..7 in elems 0..7; lane m+16 -> K8 in elem 0; rest zero.
// B: built on ALL lanes (lanes 16-31 carry K=8..15,24..31 rows; elem 0 is reset to
// cp[8] for K8, the rest multiply guaranteed-zero A entries); column = lane & 15.
// ---------------------------------------------------------------------------
__global__ void encode3x3(const float* __restrict__ act, const float* __restrict__ cent,
                          unsigned char* __restrict__ idxo, int C, int Hi, int stride) {
  __shared__ float sbuf[16 * 17];
  const int tile = blockIdx.x, c = blockIdx.y, b = blockIdx.z;
  const int lane = threadIdx.x & 31;
  const int m = lane & 15;
  const int l = tile * 16 + m;
  const int ho = l / 28, wo = l % 28;
  const float* base = act + ((size_t)(b * C + c)) * Hi * Hi;

  v16bf A = {};
  if (lane < 16) {
#pragma unroll
    for (int t = 0; t < 8; ++t) {                  // taps 0..7 -> K=0..7
      int hi = ho * stride + t / 3 - 1;
      int wi = wo * stride + t % 3 - 1;
      float v = (hi >= 0 && hi < Hi && wi >= 0 && wi < Hi) ? base[hi * Hi + wi] : 0.0f;
      A[t] = f2bf(v);
    }
  } else {                                         // tap 8 (dy=dx=2) -> K=8
    int hi = ho * stride + 1;
    int wi = wo * stride + 1;
    float v = (hi < Hi && wi < Hi) ? base[hi * Hi + wi] : 0.0f;
    A[0] = f2bf(v);
  }

  const int n = lane & 15;
  const float* cp = cent + (size_t)(c * 16 + n) * 9;
  v16bf Bf = {};
  float c2 = 0.0f;
#pragma unroll
  for (int e = 0; e < 9; ++e) { float cv = cp[e]; Bf[e] = f2bf(cv); c2 += cv * cv; }
  if (lane >= 16) Bf[0] = f2bf(cp[8]);

  wmma_argmax_store(A, Bf, c2, sbuf, idxo + ((size_t)(b * C + c)) * 784 + tile * 16);
}

// ---------------------------------------------------------------------------
// 1x1 stride-2 encode (shortcut): codebook c covers 4 consecutive channels,
// subvector d=4 = those 4 channel values at (2*ho, 2*wo). 32 codebooks.
// ---------------------------------------------------------------------------
__global__ void encode1x1(const float* __restrict__ act, const float* __restrict__ cent,
                          unsigned char* __restrict__ idxo) {
  __shared__ float sbuf[16 * 17];
  const int tile = blockIdx.x, c = blockIdx.y, b = blockIdx.z;
  const int lane = threadIdx.x & 31;
  const int m = lane & 15;
  const int l = tile * 16 + m;
  const int ho = l / 28, wo = l % 28;
  const int hi = 2 * ho, wi = 2 * wo;

  v16bf A = {};
  if (lane < 16) {
#pragma unroll
    for (int j = 0; j < 4; ++j)                    // channel offsets -> K=0..3
      A[j] = f2bf(act[(((size_t)b * 128 + c * 4 + j) * 56 + hi) * 56 + wi]);
  }

  const int n = lane & 15;
  const float* cp = cent + (size_t)(c * 16 + n) * 4;
  v16bf Bf = {};
  float c2 = 0.0f;
#pragma unroll
  for (int e = 0; e < 4; ++e) { float cv = cp[e]; Bf[e] = f2bf(cv); c2 += cv * cv; }

  wmma_argmax_store(A, Bf, c2, sbuf, idxo + ((size_t)(b * 32 + c)) * 784 + tile * 16);
}

// ---------------------------------------------------------------------------
// LUT gather-accumulate: out[b,o,l] = sum_c lut[c, idx[b,c,l], o].
// Block = 256 threads covering 16 locations x 256 channels; each thread owns
// a 4-channel x 4-location sub-tile, so every lut gather is a b128 load
// (512B contiguous per wave) and stores are b128 after a 4x4 reg transpose.
// ---------------------------------------------------------------------------
__global__ void lut_accum(const unsigned char* __restrict__ idx, const float* __restrict__ lut,
                          float* __restrict__ out, int ncb) {
  __shared__ unsigned char sidx[256 * 16];
  const int b = blockIdx.y, l0 = blockIdx.x * 16;
  const int t = threadIdx.x;
  const int og = t & 63;        // channels o = 4*og .. 4*og+3
  const int lg = t >> 6;        // locations l = l0 + 4*lg .. +3

  for (int w = t; w < ncb * 4; w += 256) {
    int c = w >> 2;
    const unsigned int* p = (const unsigned int*)(idx + ((size_t)(b * ncb + c)) * 784 + l0);
    ((unsigned int*)sidx)[w] = p[w & 3];
  }
  __syncthreads();

  float acc[4][4];              // [loc][channel]
#pragma unroll
  for (int j = 0; j < 4; ++j)
#pragma unroll
    for (int q = 0; q < 4; ++q) acc[j][q] = 0.0f;

  const float* lbase = lut + og * 4;
  for (int c = 0; c < ncb; ++c) {
    unsigned int codes = *(const unsigned int*)(sidx + c * 16 + lg * 4);
    const float* lc = lbase + (size_t)c * 4096;   // lut[c][k][o], k-stride 256
#pragma unroll
    for (int j = 0; j < 4; ++j) {
      float4 v = *(const float4*)(lc + (int)((codes >> (8 * j)) & 255u) * 256);
      acc[j][0] += v.x; acc[j][1] += v.y; acc[j][2] += v.z; acc[j][3] += v.w;
    }
  }

#pragma unroll
  for (int q = 0; q < 4; ++q) {
    float4 r = make_float4(acc[0][q], acc[1][q], acc[2][q], acc[3][q]);
    *(float4*)(out + ((size_t)(b * 256 + og * 4 + q)) * 784 + l0 + lg * 4) = r;
  }
}

// ---------------------------------------------------------------------------
// Training-mode BN over (N,H,W), two stages for memory parallelism:
// (C x 8) partial-sum blocks with b128 loads, then a tiny finalize that
// folds gamma/beta into per-channel scale/shift.
// ---------------------------------------------------------------------------
__global__ void bn_stats_part(const float* __restrict__ x, float* __restrict__ ps1,
                              float* __restrict__ ps2, int C, int HW) {
  const int c = blockIdx.x, sl = blockIdx.y, tid = threadIdx.x;
  const int HW4 = HW >> 2;
  float s1 = 0.0f, s2 = 0.0f;
  for (int b = sl * 4; b < sl * 4 + 4; ++b) {
    const float4* p = (const float4*)(x + ((size_t)(b * C + c)) * HW);
    for (int i = tid; i < HW4; i += 256) {
      float4 v = p[i];
      s1 += v.x + v.y + v.z + v.w;
      s2 += v.x * v.x + v.y * v.y + v.z * v.z + v.w * v.w;
    }
  }
  __shared__ float r1[256], r2[256];
  r1[tid] = s1; r2[tid] = s2; __syncthreads();
  for (int s = 128; s > 0; s >>= 1) {
    if (tid < s) { r1[tid] += r1[tid + s]; r2[tid] += r2[tid + s]; }
    __syncthreads();
  }
  if (tid == 0) { ps1[c * 8 + sl] = r1[0]; ps2[c * 8 + sl] = r2[0]; }
}

__global__ void bn_stats_final(const float* __restrict__ ps1, const float* __restrict__ ps2,
                               const float* __restrict__ g, const float* __restrict__ bt,
                               float* __restrict__ scale, float* __restrict__ shift,
                               int C, float inv) {
  const int c = blockIdx.x * 256 + threadIdx.x;
  if (c >= C) return;
  float s1 = 0.0f, s2 = 0.0f;
#pragma unroll
  for (int s = 0; s < 8; ++s) { s1 += ps1[c * 8 + s]; s2 += ps2[c * 8 + s]; }
  float mean = s1 * inv;
  float var  = s2 * inv - mean * mean;
  float rs   = rsqrtf(var + 1e-5f);
  float sc   = g[c] * rs;
  scale[c] = sc;
  shift[c] = bt[c] - mean * sc;
}

__global__ void bn_apply_relu(const float* __restrict__ x, const float* __restrict__ scale,
                              const float* __restrict__ shift, float* __restrict__ out,
                              int C, int HW4) {
  const int i = blockIdx.x * 256 + threadIdx.x;
  if (i >= HW4) return;
  const int c = blockIdx.y, b = blockIdx.z;
  const size_t off = ((size_t)(b * C + c)) * HW4 + i;
  const float sc = scale[c], sh = shift[c];
  float4 v = ((const float4*)x)[off];
  v.x = fmaxf(v.x * sc + sh, 0.0f);
  v.y = fmaxf(v.y * sc + sh, 0.0f);
  v.z = fmaxf(v.z * sc + sh, 0.0f);
  v.w = fmaxf(v.w * sc + sh, 0.0f);
  ((float4*)out)[off] = v;
}

// --------------------------- SE gate ---------------------------------------
__global__ void se_pool(const float* __restrict__ x, float* __restrict__ pooled) {
  const int o = blockIdx.x, b = blockIdx.y, t = threadIdx.x;
  const float4* p = (const float4*)(x + ((size_t)(b * 256 + o)) * 784);
  float s = 0.0f;
  for (int l = t; l < 196; l += 128) { float4 v = p[l]; s += v.x + v.y + v.z + v.w; }
  __shared__ float r[128];
  r[t] = s; __syncthreads();
  for (int k = 64; k > 0; k >>= 1) { if (t < k) r[t] += r[t + k]; __syncthreads(); }
  if (t == 0) pooled[b * 256 + o] = r[0] * (1.0f / 784.0f);
}

__global__ void se_fc(const float* __restrict__ pooled, const float* __restrict__ w1,
                      const float* __restrict__ b1, const float* __restrict__ w2,
                      const float* __restrict__ b2, float* __restrict__ gate) {
  const int b = blockIdx.x, t = threadIdx.x;
  __shared__ float pw[256], h[16];
  pw[t] = pooled[b * 256 + t];
  __syncthreads();
  if (t < 16) {
    float s = b1[t];
    for (int o = 0; o < 256; ++o) s += pw[o] * w1[t * 256 + o];
    h[t] = fmaxf(s, 0.0f);
  }
  __syncthreads();
  float s = b2[t];
#pragma unroll
  for (int j = 0; j < 16; ++j) s += h[j] * w2[t * 16 + j];
  gate[b * 256 + t] = 1.0f / (1.0f + expf(-s));
}

__global__ void final_combine(const float* __restrict__ out2, const float* __restrict__ gate,
                              const float* __restrict__ sc, float* __restrict__ out) {
  const int i = blockIdx.x * 256 + threadIdx.x;   // float4 index within 196
  if (i >= 196) return;
  const int o = blockIdx.y, b = blockIdx.z;
  const size_t off = ((size_t)(b * 256 + o)) * 196 + i;
  const float g = gate[b * 256 + o];
  float4 v = ((const float4*)out2)[off];
  float4 s = ((const float4*)sc)[off];
  v.x = v.x * g + s.x; v.y = v.y * g + s.y;
  v.z = v.z * g + s.z; v.w = v.w * g + s.w;
  ((float4*)out)[off] = v;
}

// ---------------------------------------------------------------------------
extern "C" void kernel_launch(void* const* d_in, const int* in_sizes, int n_in,
                              void* d_out, int out_size, void* d_ws, size_t ws_size,
                              hipStream_t stream) {
  (void)in_sizes; (void)n_in; (void)out_size; (void)ws_size;
  const float* x      = (const float*)d_in[0];
  const float* bn1_g  = (const float*)d_in[1];
  const float* bn1_b  = (const float*)d_in[2];
  const float* cent1  = (const float*)d_in[3];
  const float* lut1   = (const float*)d_in[4];
  const float* bn2_g  = (const float*)d_in[5];
  const float* bn2_b  = (const float*)d_in[6];
  const float* cent2  = (const float*)d_in[7];
  const float* lut2   = (const float*)d_in[8];
  const float* cent_s = (const float*)d_in[9];
  const float* lut_s  = (const float*)d_in[10];
  const float* fc1w   = (const float*)d_in[11];
  const float* fc1b   = (const float*)d_in[12];
  const float* fc2w   = (const float*)d_in[13];
  const float* fc2b   = (const float*)d_in[14];
  float* out = (float*)d_out;

  char* ws = (char*)d_ws;
  size_t off = 0;
  auto take = [&](size_t bytes) -> char* {
    char* p = ws + off;
    off = (off + bytes + 255) & ~(size_t)255;
    return p;
  };
  float* a1   = (float*)take((size_t)32 * 128 * 3136 * 4);   // BN1+ReLU of x
  float* sc1  = (float*)take(128 * 4);
  float* sh1  = (float*)take(128 * 4);
  float* out1 = (float*)take((size_t)32 * 256 * 784 * 4);    // conv1 result
  float* a2   = (float*)take((size_t)32 * 256 * 784 * 4);    // BN2+ReLU of out1
  float* sc2  = (float*)take(256 * 4);
  float* sh2  = (float*)take(256 * 4);
  float* scut = (float*)take((size_t)32 * 256 * 784 * 4);    // shortcut
  float* out2 = (float*)take((size_t)32 * 256 * 784 * 4);    // conv2 result
  unsigned char* idx1 = (unsigned char*)take((size_t)32 * 128 * 784);
  unsigned char* idxs = (unsigned char*)take((size_t)32 * 32 * 784);
  unsigned char* idx2 = (unsigned char*)take((size_t)32 * 256 * 784);
  float* pooled = (float*)take(32 * 256 * 4);
  float* gate   = (float*)take(32 * 256 * 4);
  float* ps1    = (float*)take(256 * 8 * 4);
  float* ps2    = (float*)take(256 * 8 * 4);

  // BN1 + ReLU   (HW=3136 -> 784 float4)
  bn_stats_part<<<dim3(128, 8), 256, 0, stream>>>(x, ps1, ps2, 128, 3136);
  bn_stats_final<<<1, 256, 0, stream>>>(ps1, ps2, bn1_g, bn1_b, sc1, sh1,
                                        128, 1.0f / (32.0f * 3136.0f));
  bn_apply_relu<<<dim3(4, 128, 32), 256, 0, stream>>>(x, sc1, sh1, a1, 128, 784);
  // conv1 (3x3 s2) + shortcut (1x1 s2): WMMA encode, then LUT accumulate
  encode3x3<<<dim3(49, 128, 32), 32, 0, stream>>>(a1, cent1, idx1, 128, 56, 2);
  encode1x1<<<dim3(49, 32, 32), 32, 0, stream>>>(a1, cent_s, idxs);
  lut_accum<<<dim3(49, 32), 256, 0, stream>>>(idx1, lut1, out1, 128);
  lut_accum<<<dim3(49, 32), 256, 0, stream>>>(idxs, lut_s, scut, 32);
  // BN2 + ReLU   (HW=784 -> 196 float4)
  bn_stats_part<<<dim3(256, 8), 256, 0, stream>>>(out1, ps1, ps2, 256, 784);
  bn_stats_final<<<1, 256, 0, stream>>>(ps1, ps2, bn2_g, bn2_b, sc2, sh2,
                                        256, 1.0f / (32.0f * 784.0f));
  bn_apply_relu<<<dim3(1, 256, 32), 256, 0, stream>>>(out1, sc2, sh2, a2, 256, 196);
  // conv2 (3x3 s1)
  encode3x3<<<dim3(49, 256, 32), 32, 0, stream>>>(a2, cent2, idx2, 256, 28, 1);
  lut_accum<<<dim3(49, 32), 256, 0, stream>>>(idx2, lut2, out2, 256);
  // SE gate + residual
  se_pool<<<dim3(256, 32), 128, 0, stream>>>(out2, pooled);
  se_fc<<<32, 256, 0, stream>>>(pooled, fc1w, fc1b, fc2w, fc2b, gate);
  final_combine<<<dim3(1, 256, 32), 256, 0, stream>>>(out2, gate, scut, out);
}
